// PathTransformerW_69200513073397
// MI455X (gfx1250) — matrix operand order repaired and checked by reading, not compiled, not run on hardware
//
#include <hip/hip_runtime.h>
#include <hip/hip_bf16.h>

#define B_ 4
#define L_ 1024
#define DIN_ 64
#define D_ 512
#define H_ 8
#define NL_ 3
#define DFF_ 2048
#define DH_ 64
#define NRBF_ 8
#define FEAT_PAD 544   // 530 padded up to a multiple of 32
#define FEAT_RAW 530
#define LARGE_NEG_ 10000.0f
#define CLAMP_ 20.0f

typedef __attribute__((ext_vector_type(16))) __bf16 v16bf;
typedef __attribute__((ext_vector_type(8)))  float  v8f;

static __device__ __forceinline__ __bf16 f2bf(float x) { return (__bf16)x; }

static __device__ __forceinline__ float geluf(float x) {
  return 0.5f * x * (1.0f + erff(x * 0.70710678118654752f));
}

static __device__ __forceinline__ v8f wmma_bf16(v16bf a, v16bf b, v8f c) {
  return __builtin_amdgcn_wmma_f32_16x16x32_bf16(false, a, false, b, (short)0, c,
                                                 false, false);
}

// pack 4 f32 -> 4 bf16, one 8-byte LDS store (dst must be 8B aligned)
static __device__ __forceinline__ void store_bf16x4(__bf16* dst, float4 f) {
  union { __bf16 h[4]; unsigned long long u; } u4;
  u4.h[0] = f2bf(f.x); u4.h[1] = f2bf(f.y);
  u4.h[2] = f2bf(f.z); u4.h[3] = f2bf(f.w);
  *(unsigned long long*)dst = u4.u;
}

// ---- WMMA fragment loaders (wave32, 16x16x32 bf16) --------------------------
// A fragment: 16(M) x 32(K). Lane l: M = l&15; VGPR pair v holds
// K = (v<4?0:16) + (v&3)*2 + ((l>=16)?8:0)  (per ISA 16-bit A table).
static __device__ __forceinline__ v16bf load_frag_A(const __bf16* base, int ld) {
  const int lane = threadIdx.x & 31;
  const int m = lane & 15;
  const int kh = (lane >> 4) << 3;
  const __bf16* row = base + m * ld;
  v16bf a;
#pragma unroll
  for (int v = 0; v < 8; ++v) {
    int kb = ((v < 4) ? 0 : 16) + ((v & 3) << 1) + kh;
    a[2 * v]     = row[kb];
    a[2 * v + 1] = row[kb + 1];
  }
  return a;
}

// B fragment: 32(K) x 16(N), column n contiguous over K in source row n.
// Lane l: N = l&15; VGPR pair v holds K = (v<4?0:16) + (v&3)*2 + ((l>=16)?8:0).
static __device__ __forceinline__ v16bf load_frag_B(const __bf16* base, int ld) {
  const int lane = threadIdx.x & 31;
  const int n = lane & 15;
  const int kh = (lane >> 4) << 3;
  const __bf16* row = base + n * ld;
  v16bf b;
#pragma unroll
  for (int v = 0; v < 8; ++v) {
    int kb = ((v < 4) ? 0 : 16) + ((v & 3) << 1) + kh;
    b[2 * v]     = row[kb];
    b[2 * v + 1] = row[kb + 1];
  }
  return b;
}

// ---- GEMM: C[M,N] = act(A[M,K] @ W[N,K]^T + bias[N]) (+R) -------------------
// Block tile 64(M) x 128(N), 8 waves; K stepped by 32 through padded bf16 LDS.
// OUTBF: emit bf16 output (used for QKV so attention streams bf16 directly).
template <int ACT, bool HAS_R, bool OUTBF>
__global__ __launch_bounds__(256) void gemm_kernel(
    const float* __restrict__ A, const float* __restrict__ W,
    const float* __restrict__ bias, const float* __restrict__ R,
    void* __restrict__ Cout, int M, int N, int K) {
  __shared__ __bf16 As[64 * 36];
  __shared__ __bf16 Ws[128 * 36];
  const int tid = threadIdx.x;
  const int lane = tid & 31;
  const int wave = tid >> 5;
  const int wm = wave & 3;
  const int wn = wave >> 2;
  const int n0 = blockIdx.x * 128;
  const int m0 = blockIdx.y * 64;

  v8f acc[4];
#pragma unroll
  for (int j = 0; j < 4; ++j) acc[j] = {};

  for (int k0 = 0; k0 < K; k0 += 32) {
#pragma unroll
    for (int i = tid; i < 512; i += 256) {   // A tile: 64 rows x 8 float4
      int r = i >> 3, c4 = i & 7;
      float4 f = *(const float4*)&A[(size_t)(m0 + r) * K + k0 + c4 * 4];
      store_bf16x4(&As[r * 36 + c4 * 4], f);
    }
#pragma unroll
    for (int i = tid; i < 1024; i += 256) {  // W tile: 128 rows x 8 float4
      int r = i >> 3, c4 = i & 7;
      float4 f = *(const float4*)&W[(size_t)(n0 + r) * K + k0 + c4 * 4];
      store_bf16x4(&Ws[r * 36 + c4 * 4], f);
    }
    __syncthreads();
    v16bf a = load_frag_A(As + wm * 16 * 36, 36);
#pragma unroll
    for (int j = 0; j < 4; ++j) {
      v16bf b = load_frag_B(Ws + (wn * 64 + j * 16) * 36, 36);
      acc[j] = wmma_bf16(a, b, acc[j]);
    }
    __syncthreads();
  }

  const int rowoff = (lane >> 4) << 3;
  const int col16 = lane & 15;
#pragma unroll
  for (int j = 0; j < 4; ++j) {
#pragma unroll
    for (int r = 0; r < 8; ++r) {
      int row = m0 + wm * 16 + r + rowoff;
      int col = n0 + wn * 64 + j * 16 + col16;
      float v = acc[j][r] + bias[col];
      if (HAS_R) v += R[(size_t)row * N + col];
      if (ACT == 1) v = geluf(v);
      if (OUTBF) ((__bf16*)Cout)[(size_t)row * N + col] = f2bf(v);
      else       ((float*)Cout)[(size_t)row * N + col] = v;
    }
  }
}

// ---- LayerNorm over D=512 (one row per block, 256 threads) ------------------
__global__ __launch_bounds__(256) void ln_kernel(const float* __restrict__ in,
                                                 const float* __restrict__ g,
                                                 const float* __restrict__ b,
                                                 float* __restrict__ out) {
  __shared__ float red[256];
  const int row = blockIdx.x;
  const int tid = threadIdx.x;
  const float* x = in + (size_t)row * D_;
  float v0 = x[tid], v1 = x[tid + 256];
  red[tid] = v0 + v1;
  __syncthreads();
  for (int o = 128; o > 0; o >>= 1) { if (tid < o) red[tid] += red[tid + o]; __syncthreads(); }
  float mean = red[0] / (float)D_;
  __syncthreads();
  float d0 = v0 - mean, d1 = v1 - mean;
  red[tid] = d0 * d0 + d1 * d1;
  __syncthreads();
  for (int o = 128; o > 0; o >>= 1) { if (tid < o) red[tid] += red[tid + o]; __syncthreads(); }
  float rstd = rsqrtf(red[0] / (float)D_ + 1e-5f);
  float* y = out + (size_t)row * D_;
  y[tid]       = d0 * rstd * g[tid]       + b[tid];
  y[tid + 256] = d1 * rstd * g[tid + 256] + b[tid + 256];
}

// h[row] += mask ? 0 : gain * LN(in[row]) * g + b
__global__ __launch_bounds__(256) void ln_pe_add_kernel(
    const float* __restrict__ in, const float* __restrict__ g,
    const float* __restrict__ b, const float* __restrict__ gain,
    const int* __restrict__ lengths, float* __restrict__ h) {
  __shared__ float red[256];
  const int row = blockIdx.x;
  const int bb = row / L_;
  const int l = row % L_;
  if (l >= lengths[bb]) return;  // masked row: pe contribution is zero
  const int tid = threadIdx.x;
  const float* x = in + (size_t)row * D_;
  float v0 = x[tid], v1 = x[tid + 256];
  red[tid] = v0 + v1;
  __syncthreads();
  for (int o = 128; o > 0; o >>= 1) { if (tid < o) red[tid] += red[tid + o]; __syncthreads(); }
  float mean = red[0] / (float)D_;
  __syncthreads();
  float d0 = v0 - mean, d1 = v1 - mean;
  red[tid] = d0 * d0 + d1 * d1;
  __syncthreads();
  for (int o = 128; o > 0; o >>= 1) { if (tid < o) red[tid] += red[tid + o]; __syncthreads(); }
  float rstd = rsqrtf(red[0] / (float)D_ + 1e-5f);
  float gn = gain[0];
  float* hr = h + (size_t)row * D_;
  hr[tid]       += gn * (d0 * rstd * g[tid]       + b[tid]);
  hr[tid + 256] += gn * (d1 * rstd * g[tid + 256] + b[tid + 256]);
}

// ---- delay encoder: Linear(1,D)->GELU->Linear(D,D)->LN ----------------------
__global__ __launch_bounds__(256) void delay_enc_kernel(
    const float* __restrict__ delay, const float* __restrict__ w1,
    const float* __restrict__ b1, const float* __restrict__ w2,
    const float* __restrict__ b2, const float* __restrict__ lng,
    const float* __restrict__ lnb, float* __restrict__ e) {
  __shared__ float t1[D_];
  __shared__ float t2[D_];
  __shared__ float red[256];
  const int bb = blockIdx.x;
  const int tid = threadIdx.x;
  float dv = delay[bb];
  for (int d = tid; d < D_; d += 256) t1[d] = geluf(dv * w1[d] + b1[d]);
  __syncthreads();
  for (int d = tid; d < D_; d += 256) {
    float acc = b2[d];
    const float* wr = w2 + (size_t)d * D_;
    for (int j = 0; j < D_; ++j) acc += t1[j] * wr[j];
    t2[d] = acc;
  }
  __syncthreads();
  red[tid] = t2[tid] + t2[tid + 256];
  __syncthreads();
  for (int o = 128; o > 0; o >>= 1) { if (tid < o) red[tid] += red[tid + o]; __syncthreads(); }
  float mean = red[0] / (float)D_;
  __syncthreads();
  float d0 = t2[tid] - mean, d1 = t2[tid + 256] - mean;
  red[tid] = d0 * d0 + d1 * d1;
  __syncthreads();
  for (int o = 128; o > 0; o >>= 1) { if (tid < o) red[tid] += red[tid + o]; __syncthreads(); }
  float rstd = rsqrtf(red[0] / (float)D_ + 1e-5f);
  e[(size_t)bb * D_ + tid]       = d0 * rstd * lng[tid]       + lnb[tid];
  e[(size_t)bb * D_ + tid + 256] = d1 * rstd * lng[tid + 256] + lnb[tid + 256];
}

// h[b,l,d] += e[b,d]
__global__ void add_e_kernel(float* __restrict__ h, const float* __restrict__ e) {
  size_t i = (size_t)blockIdx.x * blockDim.x + threadIdx.x;
  int d = (int)(i & (D_ - 1));
  int bb = (int)(i / ((size_t)L_ * D_));
  h[i] += e[(size_t)bb * D_ + d];
}

// z features: [PE(512) | cl | cs | rbf(cl)x8 | rbf(cs)x8 | pad to 544]
__global__ __launch_bounds__(256) void build_z_kernel(
    const float* __restrict__ c_local, const float* __restrict__ c_sink,
    float* __restrict__ z) {
  const int row = blockIdx.x;  // b*L + l
  const int l = row % L_;
  const int tid = threadIdx.x;
  float cl = fminf(fmaxf(c_local[row], 0.f), 1.f);
  float cs = fminf(fmaxf(c_sink[row], 0.f), 1.f);
  float* zr = z + (size_t)row * FEAT_PAD;
  for (int d = tid; d < FEAT_PAD; d += 256) {
    float v;
    if (d < D_) {
      int i = d >> 1;
      float div = __expf((float)(2 * i) * (-9.210340371976184f / (float)D_));
      float arg = (float)l * div;
      v = (d & 1) ? __cosf(arg) : __sinf(arg);
    } else if (d == D_) {
      v = cl;
    } else if (d == D_ + 1) {
      v = cs;
    } else if (d < D_ + 2 + NRBF_) {
      float c = (float)(d - (D_ + 2)) / 7.0f;
      float t = (cl - c) / 0.200001f;
      v = __expf(-0.5f * t * t);
    } else if (d < D_ + 2 + 2 * NRBF_) {
      float c = (float)(d - (D_ + 2 + NRBF_)) / 7.0f;
      float t = (cs - c) / 0.200001f;
      v = __expf(-0.5f * t * t);
    } else {
      v = 0.f;
    }
    zr[d] = v;
  }
}

__global__ void pad_w_kernel(const float* __restrict__ w, float* __restrict__ wp) {
  int i = blockIdx.x * blockDim.x + threadIdx.x;
  if (i >= D_ * FEAT_PAD) return;
  int n = i / FEAT_PAD, k = i % FEAT_PAD;
  wp[i] = (k < FEAT_RAW) ? w[n * FEAT_RAW + k] : 0.f;
}

// ---- attention bias precompute (masking folded in) --------------------------
// nbr_up[b][k]: bias for (q=k-1, k) = alpha*cl[k]               (k>=1, k<len)
// nbr_dn[b][k]: bias for (q=k+1, k), reproducing c_edge overwrite order:
//              alpha*((k==0||k==L-2)? cl[k+1] : cl[k])          (k+1<len)
// sinkv[b][k]: beta*(floor+(1-floor)*(clip(cs)+1e-6)^gamma)      (k<len)
__global__ void bias_prep_kernel(const float* __restrict__ c_local,
                                 const float* __restrict__ c_sink,
                                 const int* __restrict__ lengths,
                                 const float* __restrict__ alpha_p,
                                 const float* __restrict__ beta_p,
                                 const float* __restrict__ floor_p,
                                 const float* __restrict__ gamma_p,
                                 float* __restrict__ nbr_up,
                                 float* __restrict__ nbr_dn,
                                 float* __restrict__ sinkv) {
  int i = blockIdx.x * blockDim.x + threadIdx.x;  // B*L
  int bb = i / L_, k = i % L_;
  int len = lengths[bb];
  const float* cl = c_local + (size_t)bb * L_;
  float alpha = alpha_p[0], beta = beta_p[0], flo = floor_p[0], gam = gamma_p[0];
  float up = 0.f, dn = 0.f, sv = 0.f;
  if (k >= 1 && k < len) up = alpha * cl[k];
  if (k + 1 < len) dn = alpha * ((k == 0 || k == L_ - 2) ? cl[k + 1] : cl[k]);
  if (k < len) {
    float csv = fminf(fmaxf(c_sink[i], 0.f), 1.f);
    sv = beta * (flo + (1.f - flo) * powf(csv + 1e-6f, gam));
  }
  nbr_up[i] = up;
  nbr_dn[i] = dn;
  sinkv[i] = sv;
}

// ---- Flash-style attention, two passes (exact max for the clip semantics) ---
// Block = (b, h, 64 query rows); 4 waves, each owns a 16-row q tile.
// QKV is bf16; K/V tiles stream through LDS with 16B copies.
__global__ __launch_bounds__(128) void attn_kernel(
    const __bf16* __restrict__ qkv, const float* __restrict__ nbr_up,
    const float* __restrict__ nbr_dn, const float* __restrict__ sinkv,
    const int* __restrict__ lengths, float* __restrict__ obuf) {
  __shared__ __bf16 Ks[64 * 72];
  __shared__ __bf16 Vs[64 * 72];
  __shared__ __bf16 Pst[4 * 16 * 36];

  const int tid = threadIdx.x;
  const int lane = tid & 31;
  const int wave = tid >> 5;
  const int QT = L_ / 64;
  const int qt = blockIdx.x % QT;
  const int hh = (blockIdx.x / QT) % H_;
  const int bb = blockIdx.x / (QT * H_);
  const int m0 = qt * 64 + wave * 16;

  const int len = lengths[bb];
  const float scale = 0.125f;  // 1/sqrt(64)
  const float* bup = nbr_up + (size_t)bb * L_;
  const float* bdn = nbr_dn + (size_t)bb * L_;
  const float* bsk = sinkv + (size_t)bb * L_;

  const size_t base = (size_t)bb * L_ * (3 * D_);
  const int qoff = hh * DH_;
  const int koff = D_ + hh * DH_;
  const int voff = 2 * D_ + hh * DH_;
  const int rowoff = (lane >> 4) << 3;
  const int col16 = lane & 15;

  // Q fragments live in registers for the whole kernel (dh halves 0/1)
  v16bf qa[2];
  {
    int m = lane & 15;
    int kh = (lane >> 4) << 3;
    const __bf16* qrow = qkv + base + (size_t)(m0 + m) * (3 * D_) + qoff;
#pragma unroll
    for (int half = 0; half < 2; ++half) {
#pragma unroll
      for (int v = 0; v < 8; ++v) {
        int kb = half * 32 + ((v < 4) ? 0 : 16) + ((v & 3) << 1) + kh;
        qa[half][2 * v]     = qrow[kb];
        qa[half][2 * v + 1] = qrow[kb + 1];
      }
    }
  }

  // ---- pass 1: exact row max ----
  float rmax[8];
#pragma unroll
  for (int r = 0; r < 8; ++r) rmax[r] = -3.0e38f;

  for (int kt = 0; kt < L_; kt += 64) {
    __syncthreads();
#pragma unroll
    for (int i = tid; i < 512; i += 128) {  // 64 rows x 8 x (8 bf16 = 16B)
      int r = i >> 3, c = i & 7;
      *(uint4*)&Ks[r * 72 + c * 8] =
          *(const uint4*)&qkv[base + (size_t)(kt + r) * (3 * D_) + koff + c * 8];
    }
    __syncthreads();
#pragma unroll
    for (int sub = 0; sub < 4; ++sub) {
      v16bf kb0 = load_frag_B(Ks + (sub * 16) * 72, 72);
      v16bf kb1 = load_frag_B(Ks + (sub * 16) * 72 + 32, 72);
      v8f s = {};
      s = wmma_bf16(qa[0], kb0, s);
      s = wmma_bf16(qa[1], kb1, s);
      int kcol = kt + sub * 16 + col16;
      float up = bup[kcol], dn = bdn[kcol], sv = bsk[kcol];
      float pneg = (kcol >= len) ? -LARGE_NEG_ : 0.f;
#pragma unroll
      for (int r = 0; r < 8; ++r) {
        int q = m0 + rowoff + r;
        float badd = pneg;
        badd += (q == kcol - 1) ? up : 0.f;
        badd += (q == kcol + 1) ? dn : 0.f;
        badd += (q == 0) ? sv : 0.f;
        rmax[r] = fmaxf(rmax[r], s[r] * scale + badd);
      }
    }
  }
#pragma unroll
  for (int r = 0; r < 8; ++r) {
#pragma unroll
    for (int off = 1; off < 16; off <<= 1)
      rmax[r] = fmaxf(rmax[r], __shfl_xor(rmax[r], off, 32));
  }

  // ---- pass 2: P = exp(clip(S - max)), denom, O += P x V ----
  float dsum[8];
#pragma unroll
  for (int r = 0; r < 8; ++r) dsum[r] = 0.f;
  v8f o[4];
#pragma unroll
  for (int n = 0; n < 4; ++n) o[n] = {};

  __bf16* myP = Pst + wave * (16 * 36);

  for (int kt = 0; kt < L_; kt += 64) {
    __syncthreads();
#pragma unroll
    for (int i = tid; i < 512; i += 128) {
      int r = i >> 3, c = i & 7;
      *(uint4*)&Ks[r * 72 + c * 8] =
          *(const uint4*)&qkv[base + (size_t)(kt + r) * (3 * D_) + koff + c * 8];
      *(uint4*)&Vs[r * 72 + c * 8] =
          *(const uint4*)&qkv[base + (size_t)(kt + r) * (3 * D_) + voff + c * 8];
    }
    __syncthreads();
#pragma unroll
    for (int half = 0; half < 2; ++half) {
#pragma unroll
      for (int s2 = 0; s2 < 2; ++s2) {
        int sub = half * 2 + s2;
        v16bf kb0 = load_frag_B(Ks + (sub * 16) * 72, 72);
        v16bf kb1 = load_frag_B(Ks + (sub * 16) * 72 + 32, 72);
        v8f s = {};
        s = wmma_bf16(qa[0], kb0, s);
        s = wmma_bf16(qa[1], kb1, s);
        int kcol = kt + sub * 16 + col16;
        float up = bup[kcol], dn = bdn[kcol], sv = bsk[kcol];
        float pneg = (kcol >= len) ? -LARGE_NEG_ : 0.f;
#pragma unroll
        for (int r = 0; r < 8; ++r) {
          int q = m0 + rowoff + r;
          float badd = pneg;
          badd += (q == kcol - 1) ? up : 0.f;
          badd += (q == kcol + 1) ? dn : 0.f;
          badd += (q == 0) ? sv : 0.f;
          float x = s[r] * scale + badd - rmax[r];
          x = fminf(fmaxf(x, -CLAMP_), CLAMP_);
          float p = __expf(x);
          dsum[r] += p;
          myP[(rowoff + r) * 36 + s2 * 16 + col16] = f2bf(p);  // stage P subtile
        }
      }
      // wave-private staging: only need DS ordering within this wave
      asm volatile("s_wait_dscnt 0" ::: "memory");
      v16bf pa = load_frag_A(myP, 36);
#pragma unroll
      for (int n = 0; n < 4; ++n) {
        v16bf vb;
        int d = n * 16 + col16;
        int kh16 = (lane >> 4) << 3;
#pragma unroll
        for (int v = 0; v < 8; ++v) {
          int kl = half * 32 + ((v < 4) ? 0 : 16) + ((v & 3) << 1) + kh16;
          vb[2 * v]     = Vs[kl * 72 + d];
          vb[2 * v + 1] = Vs[(kl + 1) * 72 + d];
        }
        o[n] = wmma_bf16(pa, vb, o[n]);
      }
    }
  }

#pragma unroll
  for (int r = 0; r < 8; ++r) {
#pragma unroll
    for (int off = 1; off < 16; off <<= 1) dsum[r] += __shfl_xor(dsum[r], off, 32);
  }
#pragma unroll
  for (int n = 0; n < 4; ++n) {
#pragma unroll
    for (int r = 0; r < 8; ++r) {
      int row = m0 + rowoff + r;
      int col = hh * DH_ + n * 16 + col16;
      obuf[((size_t)bb * L_ + row) * D_ + col] = o[n][r] / dsum[r];
    }
  }
}

// ---- host orchestration -----------------------------------------------------
extern "C" void kernel_launch(void* const* d_in, const int* in_sizes, int n_in,
                              void* d_out, int out_size, void* d_ws, size_t ws_size,
                              hipStream_t stream) {
  (void)in_sizes; (void)n_in; (void)out_size; (void)ws_size;
  const float* x        = (const float*)d_in[0];
  const int*   lengths  = (const int*)d_in[1];
  const float* delay    = (const float*)d_in[2];
  const float* c_local  = (const float*)d_in[3];
  const float* c_sink   = (const float*)d_in[4];
  const float* in_w     = (const float*)d_in[5];
  const float* in_b     = (const float*)d_in[6];
  const float* de_w1    = (const float*)d_in[7];
  const float* de_b1    = (const float*)d_in[8];
  const float* de_w2    = (const float*)d_in[9];
  const float* de_b2    = (const float*)d_in[10];
  const float* de_ln_g  = (const float*)d_in[11];
  const float* de_ln_b  = (const float*)d_in[12];
  const float* cpe_w    = (const float*)d_in[13];
  const float* cpe_b    = (const float*)d_in[14];
  const float* cpe_ln_g = (const float*)d_in[15];
  const float* cpe_ln_b = (const float*)d_in[16];
  const float* gain     = (const float*)d_in[17];
  const float* alpha    = (const float*)d_in[18];
  const float* beta     = (const float*)d_in[19];
  const float* floorp   = (const float*)d_in[20];
  const float* gammap   = (const float*)d_in[21];
  const float* inproj_w = (const float*)d_in[22];
  const float* inproj_b = (const float*)d_in[23];
  const float* outproj_w= (const float*)d_in[24];
  const float* outproj_b= (const float*)d_in[25];
  const float* ln1_g    = (const float*)d_in[26];
  const float* ln1_b    = (const float*)d_in[27];
  const float* ln2_g    = (const float*)d_in[28];
  const float* ln2_b    = (const float*)d_in[29];
  const float* ff_w1    = (const float*)d_in[30];
  const float* ff_b1    = (const float*)d_in[31];
  const float* ff_w2    = (const float*)d_in[32];
  const float* ff_b2    = (const float*)d_in[33];
  const float* out_ln_g = (const float*)d_in[34];
  const float* out_ln_b = (const float*)d_in[35];

  const int M = B_ * L_;  // 4096 rows
  char* ws = (char*)d_ws;
  float*  h    = (float*)ws;  ws += (size_t)M * D_ * 4;
  float*  hn   = (float*)ws;  ws += (size_t)M * D_ * 4;
  float*  obuf = (float*)ws;  ws += (size_t)M * D_ * 4;
  __bf16* qkvb = (__bf16*)ws; ws += (size_t)M * 3 * D_ * 2;  // bf16 QKV / f32 pe_lin
  float*  big  = (float*)ws;  ws += (size_t)M * DFF_ * 4;    // z / ff1 reuse
  float*  wpad = (float*)ws;  ws += (size_t)D_ * FEAT_PAD * 4;
  float*  ebuf = (float*)ws;  ws += (size_t)B_ * D_ * 4;
  float*  bup  = (float*)ws;  ws += (size_t)B_ * L_ * 4;
  float*  bdn  = (float*)ws;  ws += (size_t)B_ * L_ * 4;
  float*  bsk  = (float*)ws;  ws += (size_t)B_ * L_ * 4;
  float* z = big;
  float* pe_lin = (float*)qkvb;  // 8MB f32 fits inside 12MB bf16 region
  float* ff1 = big;

  dim3 blk256(256), blk128(128);

  // embedding stage + bias precompute
  pad_w_kernel<<<(D_ * FEAT_PAD + 255) / 256, blk256, 0, stream>>>(cpe_w, wpad);
  build_z_kernel<<<M, blk256, 0, stream>>>(c_local, c_sink, z);
  delay_enc_kernel<<<B_, blk256, 0, stream>>>(delay, de_w1, de_b1, de_w2, de_b2,
                                              de_ln_g, de_ln_b, ebuf);
  bias_prep_kernel<<<(B_ * L_) / 256, blk256, 0, stream>>>(
      c_local, c_sink, lengths, alpha, beta, floorp, gammap, bup, bdn, bsk);
  gemm_kernel<0, false, false><<<dim3(D_ / 128, M / 64), blk256, 0, stream>>>(
      x, in_w, in_b, nullptr, h, M, D_, DIN_);
  add_e_kernel<<<(M * D_) / 256, blk256, 0, stream>>>(h, ebuf);
  gemm_kernel<0, false, false><<<dim3(D_ / 128, M / 64), blk256, 0, stream>>>(
      z, wpad, cpe_b, nullptr, pe_lin, M, D_, FEAT_PAD);
  ln_pe_add_kernel<<<M, blk256, 0, stream>>>(pe_lin, cpe_ln_g, cpe_ln_b, gain,
                                             lengths, h);

  // transformer layers
  for (int i = 0; i < NL_; ++i) {
    ln_kernel<<<M, blk256, 0, stream>>>(h, ln1_g + i * D_, ln1_b + i * D_, hn);
    gemm_kernel<0, false, true><<<dim3(3 * D_ / 128, M / 64), blk256, 0, stream>>>(
        hn, inproj_w + (size_t)i * 3 * D_ * D_, inproj_b + i * 3 * D_, nullptr,
        qkvb, M, 3 * D_, D_);
    attn_kernel<<<B_ * H_ * (L_ / 64), blk128, 0, stream>>>(
        qkvb, bup, bdn, bsk, lengths, obuf);
    gemm_kernel<0, true, false><<<dim3(D_ / 128, M / 64), blk256, 0, stream>>>(
        obuf, outproj_w + (size_t)i * D_ * D_, outproj_b + i * D_, h, h, M, D_, D_);
    ln_kernel<<<M, blk256, 0, stream>>>(h, ln2_g + i * D_, ln2_b + i * D_, hn);
    gemm_kernel<1, false, false><<<dim3(DFF_ / 128, M / 64), blk256, 0, stream>>>(
        hn, ff_w1 + (size_t)i * DFF_ * D_, ff_b1 + i * DFF_, nullptr, ff1, M, DFF_, D_);
    gemm_kernel<0, true, false><<<dim3(D_ / 128, M / 64), blk256, 0, stream>>>(
        ff1, ff_w2 + (size_t)i * D_ * DFF_, ff_b2 + i * D_, h, h, M, D_, DFF_);
  }

  ln_kernel<<<M, blk256, 0, stream>>>(h, out_ln_g, out_ln_b, (float*)d_out);
}
